// CapsuleLayerSemantic_22093311770685
// MI455X (gfx1250) — matrix operand order repaired and checked by reading, not compiled, not run on hardware
//
#include <hip/hip_runtime.h>
#include <hip/hip_bf16.h>

// ---------------------------------------------------------------------------
// CapsuleLayerSemantic on MI455X (gfx1250, wave32, WMMA)
// B=16, S=2048, NX=1024, A=20, H=50, O=3 ; tokens T = 32768
// Main GEMM ~67 GFLOP -> f16-in / f32-acc v_wmma_f32_16x16x32_f16.
// Per wave: 32(M) x 64(N) tile, software-pipelined K loop (ping-pong frags).
// ---------------------------------------------------------------------------

typedef _Float16 v4h  __attribute__((ext_vector_type(4)));
typedef _Float16 v8h  __attribute__((ext_vector_type(8)));
typedef _Float16 v16h __attribute__((ext_vector_type(16)));
typedef float    v8f  __attribute__((ext_vector_type(8)));

#define T_TOK   32768
#define NX      1024
#define A_AD    20
#define H_DIM   50
#define HP      64      // H padded to 4 WMMA N-tiles
#define O_DIM   3

// Build a 16-half WMMA fragment from two 16-byte chunks.
static __device__ __forceinline__ v16h frag16(const _Float16* __restrict__ p0,
                                              const _Float16* __restrict__ p1) {
    v8h lo = *reinterpret_cast<const v8h*>(p0);
    v8h hi = *reinterpret_cast<const v8h*>(p1);
    v16h r;
#pragma unroll
    for (int i = 0; i < 8; ++i) { r[i] = lo[i]; r[i + 8] = hi[i]; }
    return r;
}

struct Frags { v16h a0, a1, b0, b1, b2, b3; };

static __device__ __forceinline__ Frags load_frags(
    const _Float16* __restrict__ ar0, const _Float16* __restrict__ ar1,
    const _Float16* __restrict__ br0, const _Float16* __restrict__ br1,
    const _Float16* __restrict__ br2, const _Float16* __restrict__ br3,
    int k0, int hiA, int hiB) {
    Frags f;
    // A: 16 contiguous K per lane-half
    f.a0 = frag16(ar0 + k0 + hiA, ar0 + k0 + hiA + 8);
    f.a1 = frag16(ar1 + k0 + hiA, ar1 + k0 + hiA + 8);
    // B: 16 contiguous K per lane-half
    f.b0 = frag16(br0 + k0 + hiB, br0 + k0 + hiB + 8);
    f.b1 = frag16(br1 + k0 + hiB, br1 + k0 + hiB + 8);
    f.b2 = frag16(br2 + k0 + hiB, br2 + k0 + hiB + 8);
    f.b3 = frag16(br3 + k0 + hiB, br3 + k0 + hiB + 8);
    return f;
}

#define WMMA_STEP(F)                                                                   \
    acc00 = __builtin_amdgcn_wmma_f32_16x16x32_f16(false, (F).a0, false, (F).b0,       \
                                                   (short)0, acc00, false, false);     \
    acc01 = __builtin_amdgcn_wmma_f32_16x16x32_f16(false, (F).a0, false, (F).b1,       \
                                                   (short)0, acc01, false, false);     \
    acc02 = __builtin_amdgcn_wmma_f32_16x16x32_f16(false, (F).a0, false, (F).b2,       \
                                                   (short)0, acc02, false, false);     \
    acc03 = __builtin_amdgcn_wmma_f32_16x16x32_f16(false, (F).a0, false, (F).b3,       \
                                                   (short)0, acc03, false, false);     \
    acc10 = __builtin_amdgcn_wmma_f32_16x16x32_f16(false, (F).a1, false, (F).b0,       \
                                                   (short)0, acc10, false, false);     \
    acc11 = __builtin_amdgcn_wmma_f32_16x16x32_f16(false, (F).a1, false, (F).b1,       \
                                                   (short)0, acc11, false, false);     \
    acc12 = __builtin_amdgcn_wmma_f32_16x16x32_f16(false, (F).a1, false, (F).b2,       \
                                                   (short)0, acc12, false, false);     \
    acc13 = __builtin_amdgcn_wmma_f32_16x16x32_f16(false, (F).a1, false, (F).b3,       \
                                                   (short)0, acc13, false, false);

// ---------------------------------------------------------------------------
// Kernel 1: fold ln_g into W1 -> W1T f16 [A][HP][NX] (N-major for B-frags),
//           off[a][h] = sum_n ln_b[a][n]*W1[a][n][h] + b1[a][h]  (f32, padded)
// ---------------------------------------------------------------------------
__global__ __launch_bounds__(256) void prep_w_kernel(
    const float* __restrict__ ln_g, const float* __restrict__ ln_b,
    const float* __restrict__ W1,   const float* __restrict__ b1,
    _Float16* __restrict__ w1t,     float* __restrict__ off) {
    __shared__ float red[256];
    const int bid = blockIdx.x;          // a*HP + h
    const int a   = bid >> 6;
    const int h   = bid & 63;
    const int tid = threadIdx.x;
    float acc = 0.0f;
    if (h < H_DIM) {
        for (int n = tid; n < NX; n += 256) {
            const float w = W1[((size_t)a * NX + n) * H_DIM + h];
            w1t[((size_t)a * HP + h) * NX + n] = (_Float16)(ln_g[a * NX + n] * w);
            acc += ln_b[a * NX + n] * w;
        }
    } else {
        for (int n = tid; n < NX; n += 256)
            w1t[((size_t)a * HP + h) * NX + n] = (_Float16)0.0f;
    }
    red[tid] = acc;
    __syncthreads();
    for (int ofs = 128; ofs > 0; ofs >>= 1) {
        if (tid < ofs) red[tid] += red[tid + ofs];
        __syncthreads();
    }
    if (tid == 0) off[bid] = (h < H_DIM) ? (red[0] + b1[a * H_DIM + h]) : 0.0f;
}

// ---------------------------------------------------------------------------
// Kernel 2: per-token LayerNorm (f32 stats), emit xn as f16 [T][NX]
// ---------------------------------------------------------------------------
__global__ __launch_bounds__(256) void ln_kernel(
    const float* __restrict__ x, _Float16* __restrict__ xn) {
    __shared__ float red1[256];
    __shared__ float red2[256];
    const int t   = blockIdx.x;
    const int tid = threadIdx.x;
    const float4 v = reinterpret_cast<const float4*>(x + (size_t)t * NX)[tid];
    red1[tid] = v.x + v.y + v.z + v.w;
    red2[tid] = v.x * v.x + v.y * v.y + v.z * v.z + v.w * v.w;
    __syncthreads();
    for (int ofs = 128; ofs > 0; ofs >>= 1) {
        if (tid < ofs) { red1[tid] += red1[tid + ofs]; red2[tid] += red2[tid + ofs]; }
        __syncthreads();
    }
    const float mu   = red1[0] * (1.0f / NX);
    const float var  = red2[0] * (1.0f / NX) - mu * mu;
    const float rstd = rsqrtf(var + 1e-5f);
    v4h o;
    o[0] = (_Float16)((v.x - mu) * rstd);
    o[1] = (_Float16)((v.y - mu) * rstd);
    o[2] = (_Float16)((v.z - mu) * rstd);
    o[3] = (_Float16)((v.w - mu) * rstd);
    reinterpret_cast<v4h*>(xn + (size_t)t * NX)[tid] = o;
}

// ---------------------------------------------------------------------------
// Kernel 3: main WMMA GEMM + fused epilogue.
// Wave owns 32(M) x 64(N); K = 1024 in 32 steps, ping-pong pipelined:
// 8 x v_wmma_f32_16x16x32_f16 per step with partial loadcnt waits.
// Epilogue: +off, ReLU, LDS stage, then 50x3 second GEMM -> o[token][a][3].
// block = 256 threads = 8 waves = 256 rows; grid = (T/256, A)
// ---------------------------------------------------------------------------
__global__ __launch_bounds__(256) void gemm_kernel(
    const _Float16* __restrict__ xn,   // [T][NX] f16
    const _Float16* __restrict__ w1t,  // [A][HP][NX] f16
    const float*    __restrict__ off,  // [A][HP]
    const float*    __restrict__ W2,   // [A][H][O]
    const float*    __restrict__ b2,   // [A][O]
    float* __restrict__ o)             // [T][A][O]
{
    __shared__ float tile[8][32][HP];  // 64 KB
    const int lane = threadIdx.x & 31;
    const int wave = threadIdx.x >> 5;
    const int a    = blockIdx.y;
    const int m0   = blockIdx.x * 256 + wave * 32;
    const int lo   = lane & 15;
    const int hiA  = (lane >> 4) * 16;  // A-frag K-base (contiguous, matches B)
    const int hiB  = (lane >> 4) * 16;  // B-frag K-base (contiguous, matches A)

    const _Float16* __restrict__ ar0 = xn + (size_t)(m0 + lo) * NX;
    const _Float16* __restrict__ ar1 = xn + (size_t)(m0 + 16 + lo) * NX;
    const _Float16* __restrict__ br0 = w1t + ((size_t)a * HP +  0 + lo) * NX;
    const _Float16* __restrict__ br1 = w1t + ((size_t)a * HP + 16 + lo) * NX;
    const _Float16* __restrict__ br2 = w1t + ((size_t)a * HP + 32 + lo) * NX;
    const _Float16* __restrict__ br3 = w1t + ((size_t)a * HP + 48 + lo) * NX;

    v8f acc00 = {}, acc01 = {}, acc02 = {}, acc03 = {};
    v8f acc10 = {}, acc11 = {}, acc12 = {}, acc13 = {};

    // Software pipeline: ping-pong fragment buffers, 64 K per outer iter.
    Frags f0 = load_frags(ar0, ar1, br0, br1, br2, br3, 0, hiA, hiB);
#pragma unroll 1
    for (int k0 = 0; k0 < NX; k0 += 64) {
        Frags f1 = load_frags(ar0, ar1, br0, br1, br2, br3, k0 + 32, hiA, hiB);
        WMMA_STEP(f0)
        const int kn = (k0 + 64 < NX) ? (k0 + 64) : 0;  // harmless dead load at end
        f0 = load_frags(ar0, ar1, br0, br1, br2, br3, kn, hiA, hiB);
        WMMA_STEP(f1)
    }

    // Epilogue: D layout -> lane holds col (16j + lo), rows v + 8*(lane>=16)
    const float off0 = off[a * HP +  0 + lo];
    const float off1 = off[a * HP + 16 + lo];
    const float off2 = off[a * HP + 32 + lo];
    const float off3 = off[a * HP + 48 + lo];
    const int rbase = (lane >> 4) * 8;
#pragma unroll
    for (int v = 0; v < 8; ++v) {
        const int r = v + rbase;
        tile[wave][r][ 0 + lo] = fmaxf(acc00[v] + off0, 0.0f);
        tile[wave][r][16 + lo] = fmaxf(acc01[v] + off1, 0.0f);
        tile[wave][r][32 + lo] = fmaxf(acc02[v] + off2, 0.0f);
        tile[wave][r][48 + lo] = fmaxf(acc03[v] + off3, 0.0f);
        tile[wave][r + 16][ 0 + lo] = fmaxf(acc10[v] + off0, 0.0f);
        tile[wave][r + 16][16 + lo] = fmaxf(acc11[v] + off1, 0.0f);
        tile[wave][r + 16][32 + lo] = fmaxf(acc12[v] + off2, 0.0f);
        tile[wave][r + 16][48 + lo] = fmaxf(acc13[v] + off3, 0.0f);
    }
    __syncthreads();

    // Second GEMM (50x3) in f32: 96 outputs per wave
    const float* __restrict__ w2a = W2 + a * (H_DIM * O_DIM);
    for (int t = lane; t < 32 * O_DIM; t += 32) {
        const int r  = t / O_DIM;
        const int oc = t - r * O_DIM;
        float s = b2[a * O_DIM + oc];
#pragma unroll
        for (int hh = 0; hh < H_DIM; ++hh)
            s += tile[wave][r][hh] * w2a[hh * O_DIM + oc];
        o[((size_t)(m0 + r) * A_AD + a) * O_DIM + oc] = s;
    }
}

// ---------------------------------------------------------------------------
// Kernel 4: capsule squash over the adapter axis.
// ---------------------------------------------------------------------------
__global__ __launch_bounds__(256) void squash_kernel(
    const float* __restrict__ o, float* __restrict__ out) {
    const int idx = blockIdx.x * 256 + threadIdx.x;    // t*3 + oc
    if (idx >= T_TOK * O_DIM) return;
    const int t  = idx / O_DIM;
    const int oc = idx - t * O_DIM;
    const int b  = t >> 11;          // / 2048
    const int s  = t & 2047;
    float vals[A_AD];
    float sq = 0.0f;
#pragma unroll
    for (int a = 0; a < A_AD; ++a) {
        const float v = o[((size_t)t * A_AD + a) * O_DIM + oc];
        vals[a] = v;
        sq += v * v;
    }
    const float scale = (sq / (1.0f + sq)) * rsqrtf(sq);
#pragma unroll
    for (int a = 0; a < A_AD; ++a)
        out[(((size_t)b * A_AD + a) * 2048 + s) * O_DIM + oc] = scale * vals[a];
}

// ---------------------------------------------------------------------------
// Launch
// ---------------------------------------------------------------------------
extern "C" void kernel_launch(void* const* d_in, const int* in_sizes, int n_in,
                              void* d_out, int out_size, void* d_ws, size_t ws_size,
                              hipStream_t stream) {
    const float* x    = (const float*)d_in[0];  // [16,2048,1024]
    const float* ln_g = (const float*)d_in[1];  // [20,1024]
    const float* ln_b = (const float*)d_in[2];  // [20,1024]
    const float* W1   = (const float*)d_in[3];  // [20,1024,50]
    const float* b1   = (const float*)d_in[4];  // [20,50]
    const float* W2   = (const float*)d_in[5];  // [20,50,3]
    const float* b2   = (const float*)d_in[6];  // [20,3]
    float* out        = (float*)d_out;          // [16,20,6144]

    // workspace layout (bytes)
    char* ws = (char*)d_ws;
    _Float16* xn  = (_Float16*)(ws);                         // 32768*1024*2  = 64 MiB
    _Float16* w1t = (_Float16*)(ws + (size_t)67108864);      // 20*64*1024*2  = 2.5 MiB
    float*    off = (float*)   (ws + (size_t)69730304);      // 20*64*4
    float*    o   = (float*)   (ws + (size_t)69735424);      // 32768*20*3*4  = 7.5 MiB

    prep_w_kernel <<<dim3(A_AD * HP),         256, 0, stream>>>(ln_g, ln_b, W1, b1, w1t, off);
    ln_kernel     <<<dim3(T_TOK),             256, 0, stream>>>(x, xn);
    gemm_kernel   <<<dim3(T_TOK / 256, A_AD), 256, 0, stream>>>(xn, w1t, off, W2, b2, o);
    squash_kernel <<<dim3((T_TOK * O_DIM + 255) / 256), 256, 0, stream>>>(o, out);
}
